// GCN_49143015800978
// MI455X (gfx1250) — matrix operand order repaired and checked by reading, not compiled, not run on hardware
//
#include <hip/hip_runtime.h>

#define DEV_N 100000
#define DEV_E 1600000
#define DEV_D 128
#define DEV_L 3
#define DEV_G 512
#define DEV_EPS 1e-5f

typedef __attribute__((ext_vector_type(16))) __bf16 v16bf;
typedef __attribute__((ext_vector_type(8)))  float  v8f;

static __device__ __forceinline__ v8f wmma_bf16(v16bf a, v16bf b, v8f c) {
    // D(f32 16x16) = A(bf16 16x32) x B(bf16 32x16) + C
    return __builtin_amdgcn_wmma_f32_16x16x32_bf16(
        /*neg_a=*/false, a, /*neg_b=*/false, b,
        /*c_mod=*/(short)0, c, /*reuse_a=*/false, /*reuse_b=*/false);
}

// ---------------------------------------------------------------------------
// Utility kernels
// ---------------------------------------------------------------------------
__global__ void k_set_i32(int* __restrict__ p, int v, int n) {
    int i = blockIdx.x * blockDim.x + threadIdx.x;
    if (i < n) p[i] = v;
}

// histogram of edge destinations (int atomics: order-independent)
__global__ void k_hist_edges(const int* __restrict__ ei, int* __restrict__ counts) {
    int e = blockIdx.x * blockDim.x + threadIdx.x;
    if (e < DEV_E) atomicAdd(&counts[ei[DEV_E + e]], 1);
}

// histogram of batch ids
__global__ void k_hist_batch(const int* __restrict__ batch, int* __restrict__ gcounts) {
    int n = blockIdx.x * blockDim.x + threadIdx.x;
    if (n < DEV_N) atomicAdd(&gcounts[batch[n]], 1);
}

// per-block sums of counts (256 per block)
__global__ void __launch_bounds__(256) k_partial_sum(const int* __restrict__ counts,
                                                     int* __restrict__ partial, int n) {
    __shared__ int sh[256];
    int i = blockIdx.x * 256 + threadIdx.x;
    sh[threadIdx.x] = (i < n) ? counts[i] : 0;
    __syncthreads();
    for (int off = 128; off > 0; off >>= 1) {
        if (threadIdx.x < off) sh[threadIdx.x] += sh[threadIdx.x + off];
        __syncthreads();
    }
    if (threadIdx.x == 0) partial[blockIdx.x] = sh[0];
}

// single-block exclusive scan of up to 512 ints; writes total to *total
__global__ void __launch_bounds__(512) k_scan_block(const int* __restrict__ in,
                                                    int* __restrict__ out, int n,
                                                    int* __restrict__ total) {
    __shared__ int sh[512];
    int t = threadIdx.x;
    int v = (t < n) ? in[t] : 0;
    sh[t] = v;
    __syncthreads();
    for (int o = 1; o < 512; o <<= 1) {
        int u = (t >= o) ? sh[t - o] : 0;
        __syncthreads();
        sh[t] += u;
        __syncthreads();
    }
    if (t < n) out[t] = sh[t] - v;          // exclusive
    if (t == 511 && total) *total = sh[511];
}

// rowptr[i] = partial[block] + exclusive-scan-within-block(counts)
__global__ void __launch_bounds__(256) k_rowptr(const int* __restrict__ counts,
                                                const int* __restrict__ partial,
                                                int* __restrict__ rowptr, int n) {
    __shared__ int sh[256];
    int i = blockIdx.x * 256 + threadIdx.x;
    int c = (i < n) ? counts[i] : 0;
    sh[threadIdx.x] = c;
    __syncthreads();
    for (int o = 1; o < 256; o <<= 1) {
        int u = (threadIdx.x >= (unsigned)o) ? sh[threadIdx.x - o] : 0;
        __syncthreads();
        sh[threadIdx.x] += u;
        __syncthreads();
    }
    if (i < n) rowptr[i] = partial[blockIdx.x] + (sh[threadIdx.x] - c);
}

// invs[n] = rsqrt(in_degree + 1)   (GCN self-loop normalization)
__global__ void k_invs(const int* __restrict__ counts, float* __restrict__ invs, int n) {
    int i = blockIdx.x * blockDim.x + threadIdx.x;
    if (i < n) invs[i] = rsqrtf((float)counts[i] + 1.0f);
}

// fill CSR: for each edge, place src into dst's segment; also cache invs[src]
__global__ void k_fill_csr(const int* __restrict__ ei, const int* __restrict__ rowptr,
                           int* __restrict__ fillcnt, const float* __restrict__ invs,
                           int* __restrict__ csr_src, float* __restrict__ csr_w) {
    int e = blockIdx.x * blockDim.x + threadIdx.x;
    if (e >= DEV_E) return;
    int s = ei[e];
    int d = ei[DEV_E + e];
    int p = rowptr[d] + atomicAdd(&fillcnt[d], 1);
    csr_src[p] = s;
    csr_w[p] = invs[s];
}

// transpose W (k-major -> n-major) + split fp32 into bf16 hi/lo
__global__ void k_prepW(const float* __restrict__ Ws, __bf16* __restrict__ wtHi,
                        __bf16* __restrict__ wtLo) {
    int idx = blockIdx.x * blockDim.x + threadIdx.x;   // (l, n, k)
    if (idx >= DEV_L * DEV_D * DEV_D) return;
    int l = idx / (DEV_D * DEV_D);
    int r = idx % (DEV_D * DEV_D);
    int nn = r / DEV_D;
    int kk = r % DEV_D;
    float w = Ws[l * DEV_D * DEV_D + kk * DEV_D + nn];
    __bf16 hi = (__bf16)w;
    wtHi[idx] = hi;
    wtLo[idx] = (__bf16)(w - (float)hi);
}

// ---------------------------------------------------------------------------
// GEMM: hw[N,128] = hin[N,128] @ Wt^T  via 3xBF16 split WMMA (fp32-class acc)
// block = 8 waves; wave w -> 16-row strip; 8 n-tiles x 4 k-steps x 3 products
// ---------------------------------------------------------------------------
__global__ void __launch_bounds__(256) k_gemm(const float* __restrict__ hin,
                                              const __bf16* __restrict__ wtHi,
                                              const __bf16* __restrict__ wtLo,
                                              float* __restrict__ hw) {
    const int lane = threadIdx.x & 31;
    const int wave = threadIdx.x >> 5;
    const int l16  = lane & 15;
    const int half = lane >> 4;
    const int m0   = (blockIdx.x * 8 + wave) * 16;
    const int cb   = half * 8;                       // A K-chunk base per lane half

    // -------- load & split A (rows m0..m0+15, all K=128) --------
    const int arow  = m0 + l16;
    const int aload = (arow < DEV_N) ? arow : (DEV_N - 1);
    const float* abase = hin + (size_t)aload * DEV_D;

    v16bf ahi[4], alo[4];
#pragma unroll
    for (int ks = 0; ks < 4; ++ks) {
        // per ISA A layout: lane half0 -> K {0..7, 16..23}; half1 -> K {8..15, 24..31}
        const float* ap = abase + ks * 32 + cb;
        float av[16];
        float4 t0 = *(const float4*)(ap + 0);
        float4 t1 = *(const float4*)(ap + 4);
        float4 t2 = *(const float4*)(ap + 16);
        float4 t3 = *(const float4*)(ap + 20);
        av[0]=t0.x; av[1]=t0.y; av[2]=t0.z; av[3]=t0.w;
        av[4]=t1.x; av[5]=t1.y; av[6]=t1.z; av[7]=t1.w;
        av[8]=t2.x; av[9]=t2.y; av[10]=t2.z; av[11]=t2.w;
        av[12]=t3.x; av[13]=t3.y; av[14]=t3.z; av[15]=t3.w;
#pragma unroll
        for (int i = 0; i < 16; ++i) {
            __bf16 hi = (__bf16)av[i];
            ahi[ks][i] = hi;
            alo[ks][i] = (__bf16)(av[i] - (float)hi);
        }
    }

#pragma unroll
    for (int nt = 0; nt < 8; ++nt) {
        const int n = nt * 16 + l16;
        v8f acc = {0.f, 0.f, 0.f, 0.f, 0.f, 0.f, 0.f, 0.f};
#pragma unroll
        for (int ks = 0; ks < 4; ++ks) {
            // B layout: lane half0 -> K {0..7, 16..23}; half1 -> K {8..15, 24..31} of column n
            const size_t boff = (size_t)n * DEV_D + ks * 32 + half * 8;
            union { uint4 u[2]; v16bf v; } bh, bl;
            bh.u[0] = *(const uint4*)(wtHi + boff);
            bh.u[1] = *(const uint4*)(wtHi + boff + 16);
            bl.u[0] = *(const uint4*)(wtLo + boff);
            bl.u[1] = *(const uint4*)(wtLo + boff + 16);
            acc = wmma_bf16(ahi[ks], bh.v, acc);   // hi*hi
            acc = wmma_bf16(alo[ks], bh.v, acc);   // lo*hi
            acc = wmma_bf16(ahi[ks], bl.v, acc);   // hi*lo
        }
        // C/D layout: lane -> col n; VGPR j -> row m0 + half*8 + j
#pragma unroll
        for (int j = 0; j < 8; ++j) {
            const int m = m0 + half * 8 + j;
            if (m < DEV_N) hw[(size_t)m * DEV_D + n] = acc[j];
        }
    }
}

// ---------------------------------------------------------------------------
// Edge aggregation: wave per dst node (atomic-free; L2-resident gather)
// agg[d] = invs[d] * sum_{s in N(d)} hw[s]*invs[s]  +  hw[d]/deg[d] + bias
// ---------------------------------------------------------------------------
__global__ void __launch_bounds__(256) k_gather(const int* __restrict__ rowptr,
                                                const int* __restrict__ csrc,
                                                const float* __restrict__ cw,
                                                const float* __restrict__ hw,
                                                const float* __restrict__ invs,
                                                const float* __restrict__ bias,
                                                float* __restrict__ agg) {
    const int lane = threadIdx.x & 31;
    const int node = blockIdx.x * 8 + (threadIdx.x >> 5);
    if (node >= DEV_N) return;
    const int b = rowptr[node];
    const int e = rowptr[node + 1];
    float4 acc = {0.f, 0.f, 0.f, 0.f};
    for (int i = b; i < e; ++i) {
        int s = csrc[i];
        float w = cw[i];                 // invs[src]
        if (i + 1 < e)                   // gfx1250 global_prefetch_b8 of next row
            __builtin_prefetch(hw + (size_t)csrc[i + 1] * DEV_D + lane * 4, 0, 0);
        float4 v = *(const float4*)(hw + (size_t)s * DEV_D + lane * 4);
        acc.x = fmaf(v.x, w, acc.x);
        acc.y = fmaf(v.y, w, acc.y);
        acc.z = fmaf(v.z, w, acc.z);
        acc.w = fmaf(v.w, w, acc.w);
    }
    const float iv = invs[node];
    const float sc = iv * iv;            // 1/deg : self-loop coefficient
    float4 self = *(const float4*)(hw + (size_t)node * DEV_D + lane * 4);
    float4 bi   = *(const float4*)(bias + lane * 4);
    float4 o;
    o.x = fmaf(acc.x, iv, fmaf(self.x, sc, bi.x));
    o.y = fmaf(acc.y, iv, fmaf(self.y, sc, bi.y));
    o.z = fmaf(acc.z, iv, fmaf(self.z, sc, bi.z));
    o.w = fmaf(acc.w, iv, fmaf(self.w, sc, bi.w));
    *(float4*)(agg + (size_t)node * DEV_D + lane * 4) = o;
}

// ---------------------------------------------------------------------------
// BatchNorm statistics, two deterministic stages
// ---------------------------------------------------------------------------
__global__ void __launch_bounds__(128) k_stats(const float* __restrict__ agg,
                                               float* __restrict__ spart, int n) {
    const int c  = threadIdx.x;
    const int r0 = blockIdx.x * 256;
    const int r1 = (r0 + 256 < n) ? (r0 + 256) : n;
    float s = 0.f, s2 = 0.f;
    for (int r = r0; r < r1; ++r) {
        float v = agg[(size_t)r * DEV_D + c];
        s += v;
        s2 = fmaf(v, v, s2);
    }
    spart[(size_t)(blockIdx.x * 2 + 0) * DEV_D + c] = s;
    spart[(size_t)(blockIdx.x * 2 + 1) * DEV_D + c] = s2;
}

__global__ void __launch_bounds__(128) k_finalize(const float* __restrict__ spart, int nb,
                                                  const float* __restrict__ gamma,
                                                  const float* __restrict__ beta,
                                                  float* __restrict__ scsh) {
    const int c = threadIdx.x;
    float s = 0.f, s2 = 0.f;
    for (int b = 0; b < nb; ++b) {
        s  += spart[(size_t)(b * 2 + 0) * DEV_D + c];
        s2 += spart[(size_t)(b * 2 + 1) * DEV_D + c];
    }
    const float inv_n = 1.0f / (float)DEV_N;
    float mean = s * inv_n;
    float var  = fmaf(-mean, mean, s2 * inv_n);   // biased variance
    var = var > 0.f ? var : 0.f;
    float scale = gamma[c] * rsqrtf(var + DEV_EPS);
    scsh[c]           = scale;
    scsh[DEV_D + c]   = fmaf(-mean, scale, beta[c]);
}

__global__ void k_normrelu(const float* __restrict__ agg, const float* __restrict__ scsh,
                           float* __restrict__ hout) {
    int idx = blockIdx.x * blockDim.x + threadIdx.x;
    if (idx >= DEV_N * DEV_D) return;
    int c = idx & (DEV_D - 1);
    float v = fmaf(agg[idx], scsh[c], scsh[DEV_D + c]);
    hout[idx] = v > 0.f ? v : 0.f;
}

// Last layer: fused norm+relu+segmented pool. batch is sorted -> contiguous ranges.
__global__ void __launch_bounds__(128) k_pool(const float* __restrict__ agg,
                                              const float* __restrict__ scsh,
                                              const int* __restrict__ growptr,
                                              float* __restrict__ out) {
    const int g = blockIdx.x;
    const int c = threadIdx.x;
    const int n0 = growptr[g];
    const int n1 = growptr[g + 1];
    const float scale = scsh[c];
    const float shift = scsh[DEV_D + c];
    float s = 0.f;
    for (int n = n0; n < n1; ++n) {
        float v = fmaf(agg[(size_t)n * DEV_D + c], scale, shift);
        s += (v > 0.f ? v : 0.f);
    }
    out[(size_t)g * DEV_D + c] = s;
}

// ---------------------------------------------------------------------------
extern "C" void kernel_launch(void* const* d_in, const int* in_sizes, int n_in,
                              void* d_out, int out_size, void* d_ws, size_t ws_size,
                              hipStream_t stream) {
    (void)in_sizes; (void)n_in; (void)out_size; (void)ws_size;
    const float* x      = (const float*)d_in[0];
    const int*   ei     = (const int*)d_in[1];   // [2,E] row-major: src then dst
    const int*   batch  = (const int*)d_in[2];
    const float* Ws     = (const float*)d_in[3];
    const float* bs     = (const float*)d_in[4];
    const float* gammas = (const float*)d_in[5];
    const float* betas  = (const float*)d_in[6];
    float* out = (float*)d_out;

    const int NB = (DEV_N + 255) / 256;          // 391 blocks of 256 rows/counts

    char* w = (char*)d_ws;
    size_t off = 0;
    auto take = [&](size_t bytes) -> char* {
        char* p = w + off;
        off = (off + bytes + 255) & ~(size_t)255;
        return p;
    };
    float*  hw      = (float*)take((size_t)DEV_N * DEV_D * 4);
    float*  agg     = (float*)take((size_t)DEV_N * DEV_D * 4);
    float*  h       = (float*)take((size_t)DEV_N * DEV_D * 4);
    float*  invs    = (float*)take((size_t)DEV_N * 4);
    int*    counts  = (int*)take((size_t)DEV_N * 4);
    int*    fillcnt = (int*)take((size_t)DEV_N * 4);
    int*    rowptr  = (int*)take((size_t)(DEV_N + 1) * 4);
    int*    gcounts = (int*)take((size_t)DEV_G * 4);
    int*    growptr = (int*)take((size_t)(DEV_G + 1) * 4);
    int*    partial = (int*)take((size_t)512 * 4);
    int*    csr_src = (int*)take((size_t)DEV_E * 4);
    float*  csr_w   = (float*)take((size_t)DEV_E * 4);
    float*  spart   = (float*)take((size_t)NB * 2 * DEV_D * 4);
    float*  scsh    = (float*)take((size_t)2 * DEV_D * 4);
    __bf16* wtHi    = (__bf16*)take((size_t)DEV_L * DEV_D * DEV_D * 2);
    __bf16* wtLo    = (__bf16*)take((size_t)DEV_L * DEV_D * DEV_D * 2);

    // ---- one-time prep: degrees, CSR by dst, graph segments, W split ----
    k_set_i32<<<(DEV_N + 255) / 256, 256, 0, stream>>>(counts, 0, DEV_N);
    k_set_i32<<<(DEV_N + 255) / 256, 256, 0, stream>>>(fillcnt, 0, DEV_N);
    k_set_i32<<<(DEV_G + 255) / 256, 256, 0, stream>>>(gcounts, 0, DEV_G);
    k_hist_edges<<<(DEV_E + 255) / 256, 256, 0, stream>>>(ei, counts);
    k_hist_batch<<<(DEV_N + 255) / 256, 256, 0, stream>>>(batch, gcounts);
    k_partial_sum<<<NB, 256, 0, stream>>>(counts, partial, DEV_N);
    k_scan_block<<<1, 512, 0, stream>>>(partial, partial, NB, rowptr + DEV_N);
    k_rowptr<<<NB, 256, 0, stream>>>(counts, partial, rowptr, DEV_N);
    k_scan_block<<<1, 512, 0, stream>>>(gcounts, growptr, DEV_G, growptr + DEV_G);
    k_invs<<<(DEV_N + 255) / 256, 256, 0, stream>>>(counts, invs, DEV_N);
    k_fill_csr<<<(DEV_E + 255) / 256, 256, 0, stream>>>(ei, rowptr, fillcnt, invs,
                                                        csr_src, csr_w);
    k_prepW<<<(DEV_L * DEV_D * DEV_D + 255) / 256, 256, 0, stream>>>(Ws, wtHi, wtLo);

    // ---- layers ----
    for (int l = 0; l < DEV_L; ++l) {
        const float* hin = (l == 0) ? x : h;
        k_gemm<<<(DEV_N + 127) / 128, 256, 0, stream>>>(
            hin, wtHi + (size_t)l * DEV_D * DEV_D, wtLo + (size_t)l * DEV_D * DEV_D, hw);
        k_gather<<<(DEV_N + 7) / 8, 256, 0, stream>>>(rowptr, csr_src, csr_w, hw, invs,
                                                      bs + (size_t)l * DEV_D, agg);
        k_stats<<<NB, 128, 0, stream>>>(agg, spart, DEV_N);
        k_finalize<<<1, 128, 0, stream>>>(spart, NB, gammas + (size_t)l * DEV_D,
                                          betas + (size_t)l * DEV_D, scsh);
        if (l < DEV_L - 1) {
            k_normrelu<<<(DEV_N * DEV_D + 255) / 256, 256, 0, stream>>>(agg, scsh, h);
        } else {
            k_pool<<<DEV_G, 128, 0, stream>>>(agg, scsh, growptr, out);
        }
    }
}
